// Attention_8667244003382
// MI455X (gfx1250) — compile-verified, matrix operand order checked by reading, not run on hardware
//
#include <hip/hip_runtime.h>

// ---------------------------------------------------------------------------
// CDNA5 (gfx1250) multi-head attention forward, bf16 WMMA, fp32 accumulate.
// B=8, N=1024, C=1024, H=16, d=64, SCALE = 1/8.
//  - GEMMs: 32x64 wave tiles; shared W panel staged via async global->LDS
//    copies (double buffered on ASYNCcnt); A direct from global.
//  - Attention: 4 waves/block share one head; K/V tiles staged via async
//    global->LDS copies (double buffered); flash-style online softmax.
// ---------------------------------------------------------------------------

typedef __bf16 v16bf __attribute__((ext_vector_type(16)));
typedef float  v8f   __attribute__((ext_vector_type(8)));

#define WMMA_BF16(a, b, c) \
    __builtin_amdgcn_wmma_f32_16x16x32_bf16(false, (a), false, (b), (short)0, (c), false, false)

static __device__ __forceinline__ v16bf ldg16(const __bf16* p) {
    return *(const v16bf*)p;
}
static __device__ __forceinline__ v16bf lds16(const __bf16* p) {
    return *(const v16bf*)p;
}

// Async global -> LDS copy of 16 bytes per lane (gfx1250, tracked by ASYNCcnt).
static __device__ __forceinline__ void async_copy_b128(unsigned lds_off, const void* gptr) {
    asm volatile("global_load_async_to_lds_b128 %0, %1, off"
                 :: "v"(lds_off), "v"(gptr) : "memory");
}
static __device__ __forceinline__ void wait_async0() {
    asm volatile("s_wait_asynccnt 0" ::: "memory");
}
// Low 32 bits of a generic LDS pointer == wave-relative LDS byte address.
static __device__ __forceinline__ unsigned lds_addr(const void* p) {
    return (unsigned)(unsigned long long)p;
}

// ---------------------------------------------------------------------------
// fp32 -> bf16 conversion (grid-stride)
// ---------------------------------------------------------------------------
__global__ void k_f32_to_bf16(const float* __restrict__ s, __bf16* __restrict__ d, int n) {
    int i = blockIdx.x * blockDim.x + threadIdx.x;
    int stride = gridDim.x * blockDim.x;
    for (; i < n; i += stride) d[i] = (__bf16)s[i];
}

// ---------------------------------------------------------------------------
// One GEMM K-step: 8 WMMAs on a 32(M) x 64(N) wave tile, B from LDS tile.
// ---------------------------------------------------------------------------
static __device__ __forceinline__ void gemm_step(
        const __bf16* __restrict__ cw, const __bf16* __restrict__ aBase0,
        const __bf16* __restrict__ aBase1, int k0, int lr, int lh,
        v8f& c00, v8f& c01, v8f& c02, v8f& c03,
        v8f& c10, v8f& c11, v8f& c12, v8f& c13)
{
    v16bf a0 = ldg16(aBase0 + k0);
    v16bf a1 = ldg16(aBase1 + k0);
    v16bf b0 = lds16(&cw[(lr)      * 32 + lh * 16]);
    v16bf b1 = lds16(&cw[(16 + lr) * 32 + lh * 16]);
    v16bf b2 = lds16(&cw[(32 + lr) * 32 + lh * 16]);
    v16bf b3 = lds16(&cw[(48 + lr) * 32 + lh * 16]);
    c00 = WMMA_BF16(a0, b0, c00);
    c10 = WMMA_BF16(a1, b0, c10);
    c01 = WMMA_BF16(a0, b1, c01);
    c11 = WMMA_BF16(a1, b1, c11);
    c02 = WMMA_BF16(a0, b2, c02);
    c12 = WMMA_BF16(a1, b2, c12);
    c03 = WMMA_BF16(a0, b3, c03);
    c13 = WMMA_BF16(a1, b3, c13);
}

// Main loop shared by both GEMMs.  acc[half][s] filled on return.
static __device__ __forceinline__ void gemm_mainloop(
        const __bf16* __restrict__ aBase0, const __bf16* __restrict__ aBase1,
        const __bf16* __restrict__ wPanel,
        __bf16* __restrict__ wt0, __bf16* __restrict__ wt1,
        int tid, int lr, int lh, v8f acc[2][4])
{
    const int K = 1024;
    const int c0 = tid, c1 = tid + 128;   // 256 x 16B chunks per 4KB tile
    const char* g0 = (const char*)wPanel + (size_t)(c0 >> 2) * (K * 2) + (c0 & 3) * 16;
    const char* g1 = (const char*)wPanel + (size_t)(c1 >> 2) * (K * 2) + (c1 & 3) * 16;
    const unsigned l0 = lds_addr(wt0) + c0 * 16, l0b = lds_addr(wt0) + c1 * 16;
    const unsigned l1 = lds_addr(wt1) + c0 * 16, l1b = lds_addr(wt1) + c1 * 16;

    v8f c00 = {}, c01 = {}, c02 = {}, c03 = {};
    v8f c10 = {}, c11 = {}, c12 = {}, c13 = {};

    async_copy_b128(l0, g0);
    async_copy_b128(l0b, g1);

    for (int j = 0; j < 32; j += 2) {
        wait_async0();
        __syncthreads();
        {   // stage k-step j+1 while computing j
            const int kb = (j + 1) * 64;
            async_copy_b128(l1, g0 + kb);
            async_copy_b128(l1b, g1 + kb);
        }
        gemm_step(wt0, aBase0, aBase1, j * 32, lr, lh,
                  c00, c01, c02, c03, c10, c11, c12, c13);

        wait_async0();
        __syncthreads();
        if (j + 2 < 32) {
            const int kb = (j + 2) * 64;
            async_copy_b128(l0, g0 + kb);
            async_copy_b128(l0b, g1 + kb);
        }
        gemm_step(wt1, aBase0, aBase1, (j + 1) * 32, lr, lh,
                  c00, c01, c02, c03, c10, c11, c12, c13);
    }

    acc[0][0] = c00; acc[0][1] = c01; acc[0][2] = c02; acc[0][3] = c03;
    acc[1][0] = c10; acc[1][1] = c11; acc[1][2] = c12; acc[1][3] = c13;
}

// ---------------------------------------------------------------------------
// QKV projection:  qkv[m, n] = X[m, :] . Wq[n, :] + bias[n]
// Scatters into Q [B,H,N,64] (scaled by 1/8), K [B,H,N,64], Vt [B,H,64,N].
// grid: 3072 blocks x 128  (mBlock = bx & 63, tileN = bx >> 6)
// ---------------------------------------------------------------------------
__global__ void __launch_bounds__(128) k_qkv_gemm(
        const __bf16* __restrict__ X, const __bf16* __restrict__ W,
        const float* __restrict__ bias,
        __bf16* __restrict__ Q, __bf16* __restrict__ Kd, __bf16* __restrict__ Vt)
{
    __shared__ __align__(32) __bf16 wt[2][64 * 32];
    const int tid  = threadIdx.x;
    const int lane = tid & 31;
    const int wid  = tid >> 5;
    const int tileN = blockIdx.x >> 6;             // 48 N-panels of 64
    const int tileM = (blockIdx.x & 63) * 4 + wid; // 256 M-tiles of 32
    const int lr = lane & 15, lh = lane >> 4;
    const int K = 1024;

    const __bf16* aBase0 = X + (size_t)(tileM * 32 + lr) * K + lh * 16;
    const __bf16* aBase1 = aBase0 + (size_t)16 * K;
    const __bf16* wPanel = W + (size_t)(tileN * 64) * K;

    v8f acc[2][4];
    gemm_mainloop(aBase0, aBase1, wPanel, &wt[0][0], &wt[1][0], tid, lr, lh, acc);

    #pragma unroll
    for (int half = 0; half < 2; half++) {
        #pragma unroll
        for (int s = 0; s < 4; s++) {
            const int n   = tileN * 64 + s * 16 + lr;   // [0, 3072)
            const int sel = n >> 10;                    // 0=q 1=k 2=v
            const int h   = (n >> 6) & 15;
            const int dd  = n & 63;
            const float bv = bias[n];
            #pragma unroll
            for (int r = 0; r < 8; r++) {
                const int m    = tileM * 32 + half * 16 + lh * 8 + r;
                const int bidx = m >> 10;
                const int nq   = m & 1023;
                const size_t bh = (size_t)(bidx * 16 + h);
                float val = acc[half][s][r] + bv;
                if (sel == 0) {
                    Q[(bh * 1024 + nq) * 64 + dd] = (__bf16)(val * 0.125f);
                } else if (sel == 1) {
                    Kd[(bh * 1024 + nq) * 64 + dd] = (__bf16)val;
                } else {
                    Vt[(bh * 64 + dd) * 1024 + nq] = (__bf16)val;
                }
            }
        }
    }
}

// ---------------------------------------------------------------------------
// Flash attention.  4 waves/block, same head, 4 consecutive 16-row q-tiles.
// K tile (32 keys x 64d, 4KB) and V tile (64d x 32 keys, 4KB) staged via
// async copies, double buffered.  grid: 2048 blocks x 128.
// ---------------------------------------------------------------------------
struct AttnState {
    v8f o0, o1, o2, o3;
    float mrow[8], lrow[8];
};

static __device__ __forceinline__ void attn_step(
        const __bf16* __restrict__ kb, const __bf16* __restrict__ vb,
        __bf16* __restrict__ pb, const v16bf qf0, const v16bf qf1,
        int lr, int lh, AttnState& st)
{
    v8f s0 = {}, s1 = {};
    v16bf k00 = lds16(&kb[(lr)      * 64 + lh * 16]);
    v16bf k01 = lds16(&kb[(lr)      * 64 + 32 + lh * 16]);
    v16bf k10 = lds16(&kb[(16 + lr) * 64 + lh * 16]);
    v16bf k11 = lds16(&kb[(16 + lr) * 64 + 32 + lh * 16]);
    s0 = WMMA_BF16(qf0, k00, s0);
    s0 = WMMA_BF16(qf1, k01, s0);
    s1 = WMMA_BF16(qf0, k10, s1);
    s1 = WMMA_BF16(qf1, k11, s1);

    #pragma unroll
    for (int r = 0; r < 8; r++) {
        float a = s0[r], b = s1[r];
        float mt = fmaxf(a, b);
        mt = fmaxf(mt, __shfl_xor(mt, 1, 16));
        mt = fmaxf(mt, __shfl_xor(mt, 2, 16));
        mt = fmaxf(mt, __shfl_xor(mt, 4, 16));
        mt = fmaxf(mt, __shfl_xor(mt, 8, 16));
        float mnew = fmaxf(st.mrow[r], mt);
        float corr = __expf(st.mrow[r] - mnew);
        float p0 = __expf(a - mnew);
        float p1 = __expf(b - mnew);
        float sum = p0 + p1;
        sum += __shfl_xor(sum, 1, 16);
        sum += __shfl_xor(sum, 2, 16);
        sum += __shfl_xor(sum, 4, 16);
        sum += __shfl_xor(sum, 8, 16);
        st.lrow[r] = st.lrow[r] * corr + sum;
        st.mrow[r] = mnew;
        st.o0[r] *= corr; st.o1[r] *= corr; st.o2[r] *= corr; st.o3[r] *= corr;
        const int row = lh * 8 + r;
        pb[row * 32 + lr]      = (__bf16)p0;
        pb[row * 32 + 16 + lr] = (__bf16)p1;
    }

    v16bf pf = lds16(&pb[lr * 32 + lh * 16]);
    v16bf v0 = lds16(&vb[(lr)      * 32 + lh * 16]);
    v16bf v1 = lds16(&vb[(16 + lr) * 32 + lh * 16]);
    v16bf v2 = lds16(&vb[(32 + lr) * 32 + lh * 16]);
    v16bf v3 = lds16(&vb[(48 + lr) * 32 + lh * 16]);
    st.o0 = WMMA_BF16(pf, v0, st.o0);
    st.o1 = WMMA_BF16(pf, v1, st.o1);
    st.o2 = WMMA_BF16(pf, v2, st.o2);
    st.o3 = WMMA_BF16(pf, v3, st.o3);
}

__global__ void __launch_bounds__(128) k_attn(
        const __bf16* __restrict__ Q, const __bf16* __restrict__ Kd,
        const __bf16* __restrict__ Vt, __bf16* __restrict__ Z)
{
    __shared__ __align__(32) __bf16 kbuf[2][32 * 64];
    __shared__ __align__(32) __bf16 vbuf[2][64 * 32];
    __shared__ __align__(32) __bf16 pbuf[4][16 * 32];
    const int tid  = threadIdx.x;
    const int lane = tid & 31;
    const int wid  = tid >> 5;
    const int bh = blockIdx.x >> 4;               // 128 heads (b*16+h)
    const int qt = (blockIdx.x & 15) * 4 + wid;   // 64 q-tiles per head
    const int lr = lane & 15, lh = lane >> 4;

    const size_t headQK = (size_t)bh * 1024 * 64;
    const size_t headV  = (size_t)bh * 64 * 1024;

    // Cooperative copy plan: K tile rows are 128B (8 chunks), V tile rows 64B
    // of a 2048B-stride row.  256 chunks each; 2 per thread per tile.
    const int c0 = tid, c1 = tid + 128;
    const char* kgBase = (const char*)(Kd + headQK);
    const char* vgBase = (const char*)(Vt + headV);
    const char* kg0 = kgBase + (size_t)(c0 >> 3) * 128 + (c0 & 7) * 16;
    const char* kg1 = kgBase + (size_t)(c1 >> 3) * 128 + (c1 & 7) * 16;
    const char* vg0 = vgBase + (size_t)(c0 >> 2) * 2048 + (c0 & 3) * 16;
    const char* vg1 = vgBase + (size_t)(c1 >> 2) * 2048 + (c1 & 3) * 16;
    const unsigned lk0 = lds_addr(&kbuf[0][0]), lk1 = lds_addr(&kbuf[1][0]);
    const unsigned lv0 = lds_addr(&vbuf[0][0]), lv1 = lds_addr(&vbuf[1][0]);

    const __bf16* qbase = Q + headQK + (size_t)(qt * 16 + lr) * 64 + lh * 16;
    const v16bf qf0 = ldg16(qbase);
    const v16bf qf1 = ldg16(qbase + 32);

    AttnState st;
    st.o0 = {}; st.o1 = {}; st.o2 = {}; st.o3 = {};
    #pragma unroll
    for (int r = 0; r < 8; r++) { st.mrow[r] = -1e30f; st.lrow[r] = 0.f; }
    __bf16* pb = &pbuf[wid][0];

    // prologue: stage key-block 0 into buffer 0
    async_copy_b128(lk0 + c0 * 16, kg0);
    async_copy_b128(lk0 + c1 * 16, kg1);
    async_copy_b128(lv0 + c0 * 16, vg0);
    async_copy_b128(lv0 + c1 * 16, vg1);

    for (int j = 0; j < 32; j += 2) {
        wait_async0();
        __syncthreads();
        {   // stage key-block j+1 (K: +4096B/block, V: +64B/block)
            const size_t ko = (size_t)(j + 1) * 4096, vo = (size_t)(j + 1) * 64;
            async_copy_b128(lk1 + c0 * 16, kg0 + ko);
            async_copy_b128(lk1 + c1 * 16, kg1 + ko);
            async_copy_b128(lv1 + c0 * 16, vg0 + vo);
            async_copy_b128(lv1 + c1 * 16, vg1 + vo);
        }
        attn_step(&kbuf[0][0], &vbuf[0][0], pb, qf0, qf1, lr, lh, st);

        wait_async0();
        __syncthreads();
        if (j + 2 < 32) {
            const size_t ko = (size_t)(j + 2) * 4096, vo = (size_t)(j + 2) * 64;
            async_copy_b128(lk0 + c0 * 16, kg0 + ko);
            async_copy_b128(lk0 + c1 * 16, kg1 + ko);
            async_copy_b128(lv0 + c0 * 16, vg0 + vo);
            async_copy_b128(lv0 + c1 * 16, vg1 + vo);
        }
        attn_step(&kbuf[1][0], &vbuf[1][0], pb, qf0, qf1, lr, lh, st);
    }

    const int b = bh >> 4, h = bh & 15;
    v8f oo[4] = {st.o0, st.o1, st.o2, st.o3};
    #pragma unroll
    for (int r = 0; r < 8; r++) {
        const float inv = 1.0f / st.lrow[r];
        const int row = qt * 16 + lh * 8 + r;
        const size_t base = ((size_t)(b * 1024 + row)) * 1024 + h * 64;
        #pragma unroll
        for (int t = 0; t < 4; t++) {
            Z[base + t * 16 + lr] = (__bf16)(oo[t][r] * inv);
        }
    }
}

// ---------------------------------------------------------------------------
// Output projection: out[m, n] = Z[m, :] . Wp[n, :] + proj_b[n]  (fp32 out)
// grid: 1024 blocks x 128  (mBlock = bx & 63, tileN = bx >> 6)
// ---------------------------------------------------------------------------
__global__ void __launch_bounds__(128) k_proj_gemm(
        const __bf16* __restrict__ Zb, const __bf16* __restrict__ W,
        const float* __restrict__ bias, float* __restrict__ out)
{
    __shared__ __align__(32) __bf16 wt[2][64 * 32];
    const int tid  = threadIdx.x;
    const int lane = tid & 31;
    const int wid  = tid >> 5;
    const int tileN = blockIdx.x >> 6;              // 16 N-panels of 64
    const int tileM = (blockIdx.x & 63) * 4 + wid;  // 256 M-tiles of 32
    const int lr = lane & 15, lh = lane >> 4;
    const int K = 1024;

    const __bf16* aBase0 = Zb + (size_t)(tileM * 32 + lr) * K + lh * 16;
    const __bf16* aBase1 = aBase0 + (size_t)16 * K;
    const __bf16* wPanel = W + (size_t)(tileN * 64) * K;

    v8f acc[2][4];
    gemm_mainloop(aBase0, aBase1, wPanel, &wt[0][0], &wt[1][0], tid, lr, lh, acc);

    #pragma unroll
    for (int half = 0; half < 2; half++) {
        #pragma unroll
        for (int s = 0; s < 4; s++) {
            const int n = tileN * 64 + s * 16 + lr;
            const float bv = bias[n];
            #pragma unroll
            for (int r = 0; r < 8; r++) {
                const int m = tileM * 32 + half * 16 + lh * 8 + r;
                out[(size_t)m * 1024 + n] = acc[half][s][r] + bv;
            }
        }
    }
}

// ---------------------------------------------------------------------------
// Launcher. Workspace layout (bytes):
//   Xb   @ 0        16 MB   x as bf16            [8192,1024]
//   Wqkv @ 16 MB     6 MB   qkv_w bf16           [3072,1024]
//   Wp   @ 22 MB     2 MB   proj_w bf16          [1024,1024]
//   Qb   @ 24 MB    16 MB   q (scaled) bf16      [B,H,1024,64]
//   Kb   @ 40 MB    16 MB   k bf16               [B,H,1024,64]
//   Vt   @ 56 MB    16 MB   v^T bf16             [B,H,64,1024]
//   Zb   @ 72 MB    16 MB   attention out bf16   [8192,1024]
// Total 88 MB.
// ---------------------------------------------------------------------------
extern "C" void kernel_launch(void* const* d_in, const int* in_sizes, int n_in,
                              void* d_out, int out_size, void* d_ws, size_t ws_size,
                              hipStream_t stream) {
    (void)in_sizes; (void)n_in; (void)out_size; (void)ws_size;
    const float* x      = (const float*)d_in[0];
    const float* qkv_w  = (const float*)d_in[1];
    const float* qkv_b  = (const float*)d_in[2];
    const float* proj_w = (const float*)d_in[3];
    const float* proj_b = (const float*)d_in[4];
    float* out = (float*)d_out;

    char* ws = (char*)d_ws;
    __bf16* Xb = (__bf16*)(ws);
    __bf16* Wq = (__bf16*)(ws + (16ull << 20));
    __bf16* Wp = (__bf16*)(ws + (22ull << 20));
    __bf16* Qb = (__bf16*)(ws + (24ull << 20));
    __bf16* Kb = (__bf16*)(ws + (40ull << 20));
    __bf16* Vt = (__bf16*)(ws + (56ull << 20));
    __bf16* Zb = (__bf16*)(ws + (72ull << 20));

    k_f32_to_bf16<<<2048, 256, 0, stream>>>(x,      Xb, 8 * 1024 * 1024);
    k_f32_to_bf16<<<1024, 256, 0, stream>>>(qkv_w,  Wq, 3 * 1024 * 1024);
    k_f32_to_bf16<<<512,  256, 0, stream>>>(proj_w, Wp, 1024 * 1024);

    k_qkv_gemm <<<3072, 128, 0, stream>>>(Xb, Wq, qkv_b, Qb, Kb, Vt);
    k_attn     <<<2048, 128, 0, stream>>>(Qb, Kb, Vt, Zb);
    k_proj_gemm<<<1024, 128, 0, stream>>>(Zb, Wp, proj_b, out);
}
